// SpatialInteraction_40097814675819
// MI455X (gfx1250) — matrix operand order repaired and checked by reading, not compiled, not run on hardware
//
#include <hip/hip_runtime.h>
#include <math.h>

typedef _Float16 half_t;
typedef __attribute__((ext_vector_type(16))) _Float16 v16h;
typedef __attribute__((ext_vector_type(8)))  _Float16 v8h;
typedef __attribute__((ext_vector_type(8)))  float    v8f;

union V16 { v16h v; v8h h[2]; };

constexpr int Bn = 4, Tn = 20, Cn = 128, Dn = 64;
constexpr int TD = Tn * Dn;          // 1280
constexpr int H1 = 128, H2 = 256, H3 = 64;

// workspace layout (bytes)
constexpr size_t OFF_FU  = 0;                         // B*C*H1 f32 = 256 KB
constexpr size_t OFF_FV  = 256 * 1024;                // 256 KB
constexpr size_t OFF_W   = 512 * 1024;                // B*C*C f32 = 256 KB
constexpr size_t OFF_W2H = 768 * 1024;                // 256*128 f16 = 64 KB
constexpr size_t OFF_W3H = OFF_W2H + 64 * 1024;       // 64*256 f16 = 32 KB
constexpr size_t OFF_P   = OFF_W3H + 32 * 1024;       // B*T*C*D f32 = 2.5 MB
constexpr size_t OFF_Q   = OFF_P + (size_t)Bn*Tn*Cn*Dn*4;

// k2 dynamic-LDS partition (bytes); padded strides (in halves) kill bank conflicts
constexpr int X1S = 136;   // x1 row stride   (128 data + 8 pad)
constexpr int STS = 40;    // stage row stride (32 data + 8 pad)
constexpr int W2SS = 136;  // W2 LDS row stride
constexpr int W3SS = 264;  // W3 LDS row stride
constexpr size_t SM_X1 = 0;                               // 8*16*136*2 = 34816
constexpr size_t SM_ST = 34816;                           // 8*16*40*2  = 10240
constexpr size_t SM_W2 = 45056;                           // 256*136*2  = 69632
constexpr size_t SM_W3 = 114688;                          // 64*264*2   = 33792
constexpr size_t K2_SMEM = 148480;                        // 145 KB of 320 KB/WGP

static __device__ __forceinline__ void compiler_fence() {
  __asm__ volatile("" ::: "memory");
}

// ---------------------------------------------------------------- k0: f32->f16 weights
__global__ void k0_convert(const float* __restrict__ W2, const float* __restrict__ W3,
                           half_t* __restrict__ W2h, half_t* __restrict__ W3h) {
  int gid = blockIdx.x * blockDim.x + threadIdx.x;
  if (gid < H2 * H1) W2h[gid] = (half_t)W2[gid];
  if (gid < H3 * H2) W3h[gid] = (half_t)W3[gid];
}

// ---------------------------------------------------------------- k1: fu/fv projections
// 64 blocks: each owns (b, 8 columns c). W1 rows read coalesced across lanes, reused 8x.
__global__ void __launch_bounds__(256)
k1_fufv(const float* __restrict__ h, const float* __restrict__ W1,
        float* __restrict__ fu, float* __restrict__ fv) {
  __shared__ float hrows[8][TD];               // 40 KB
  int b = blockIdx.x >> 4;
  int c0 = (blockIdx.x & 15) * 8;
  for (int idx = threadIdx.x; idx < 8 * TD; idx += 256) {
    int ci = idx / TD, k = idx - ci * TD;
    int t = k >> 6, d = k & 63;
    hrows[ci][k] = h[(((size_t)b * Tn + t) * Cn + (c0 + ci)) * Dn + d];
  }
  __syncthreads();
  int lane = threadIdx.x & 31, wvid = threadIdx.x >> 5;
  for (int j = 0; j < 16; ++j) {
    int m = wvid + 8 * j;                      // 8 waves x 16 -> all 128 outputs
    const float* wu = W1 + (size_t)m * (2 * TD);
    const float* wv = wu + TD;
    float pu[8] = {0.f,0.f,0.f,0.f,0.f,0.f,0.f,0.f};
    float pv[8] = {0.f,0.f,0.f,0.f,0.f,0.f,0.f,0.f};
    for (int k = lane; k < TD; k += 32) {      // coalesced W1 reads
      float a = wu[k], bb = wv[k];
#pragma unroll
      for (int ci = 0; ci < 8; ++ci) {
        float hv = hrows[ci][k];
        pu[ci] = fmaf(a, hv, pu[ci]);
        pv[ci] = fmaf(bb, hv, pv[ci]);
      }
    }
#pragma unroll
    for (int msk = 1; msk < 32; msk <<= 1) {
#pragma unroll
      for (int ci = 0; ci < 8; ++ci) {
        pu[ci] += __shfl_xor(pu[ci], msk, 32);
        pv[ci] += __shfl_xor(pv[ci], msk, 32);
      }
    }
    if (lane == 0) {
#pragma unroll
      for (int ci = 0; ci < 8; ++ci) {
        fu[((size_t)b * Cn + c0 + ci) * H1 + m] = pu[ci];
        fv[((size_t)b * Cn + c0 + ci) * H1 + m] = pv[ci];
      }
    }
  }
}

// ---------------------------------------------------------------- k2: pairwise MLP (WMMA)
// grid: 4 (b) * 8 (u-tiles of 16) * 16 (v-tiles of 8) = 512 blocks, 256 threads (8 waves).
// W2h/W3h staged once per workgroup into LDS; inner loop is pure LDS + WMMA.
__global__ void __launch_bounds__(256)
k2_pairmlp(const float* __restrict__ fu, const float* __restrict__ fv,
           const float* __restrict__ b1, const half_t* __restrict__ W2h,
           const float* __restrict__ b2, const half_t* __restrict__ W3h,
           const float* __restrict__ b3, const float* __restrict__ W4,
           const float* __restrict__ b4, const float* __restrict__ A_a,
           float* __restrict__ wout) {
  extern __shared__ __align__(16) char smem[];
  half_t* x1h = (half_t*)(smem + SM_X1);   // [8][16][X1S]
  half_t* stg = (half_t*)(smem + SM_ST);   // [8][16][STS]
  half_t* W2s = (half_t*)(smem + SM_W2);   // [256][W2SS]
  half_t* W3s = (half_t*)(smem + SM_W3);   // [64][W3SS]

  int blk = blockIdx.x;
  int b  = blk >> 7;
  int u0 = ((blk & 127) >> 4) * 16;
  int v0 = (blk & 15) * 8;

  // stage weights into LDS (coalesced 16B chunks)
  for (int idx = threadIdx.x; idx < 256 * 16 + 64 * 32; idx += 256) {
    if (idx < 4096) {
      int row = idx >> 4, seg = idx & 15;
      *reinterpret_cast<v8h*>(&W2s[row * W2SS + seg * 8]) =
          *reinterpret_cast<const v8h*>(&W2h[row * H1 + seg * 8]);
    } else {
      int j = idx - 4096;
      int row = j >> 5, seg = j & 31;
      *reinterpret_cast<v8h*>(&W3s[row * W3SS + seg * 8]) =
          *reinterpret_cast<const v8h*>(&W3h[row * H2 + seg * 8]);
    }
  }
  // build x1 = relu(fu[u] + fv[v] + b1) in f16
  for (int idx = threadIdx.x; idx < 8 * 16 * H1; idx += 256) {
    int w = idx >> 11, m = (idx >> 7) & 15, k = idx & 127;
    float val = fu[((size_t)b * Cn + u0 + m) * H1 + k]
              + fv[((size_t)b * Cn + v0 + w) * H1 + k] + b1[k];
    x1h[(w * 16 + m) * X1S + k] = (half_t)fmaxf(val, 0.f);
  }
  __syncthreads();

  const int lane = threadIdx.x & 31;
  const int wv   = threadIdx.x >> 5;     // wave id -> v = v0 + wv
  const int m16  = lane & 15;
  const int hi16 = lane >> 4;

  // A operand (16x32 f16), ISA striping: lanes0-15 K{0-7,16-23}, lanes16-31 K{8-15,24-31}
  const int abase = (wv * 16 + m16) * X1S;
  V16 A1[4];
#pragma unroll
  for (int c = 0; c < 4; ++c) {
    A1[c].h[0] = *reinterpret_cast<const v8h*>(&x1h[abase + 32 * c + 8 * hi16]);
    A1[c].h[1] = *reinterpret_cast<const v8h*>(&x1h[abase + 32 * c + 16 + 8 * hi16]);
  }

  v8f zf = {0.f, 0.f, 0.f, 0.f, 0.f, 0.f, 0.f, 0.f};
  v8f acc3[4] = {zf, zf, zf, zf};

  for (int cc = 0; cc < 8; ++cc) {        // 32 layer-2 columns per pass
    v8f acc2[2] = {zf, zf};
#pragma unroll
    for (int c = 0; c < 4; ++c) {         // K = 128 over layer 2
#pragma unroll
      for (int nt = 0; nt < 2; ++nt) {
        int row = (2 * cc + nt) * 16 + m16;
        V16 Bop;
        Bop.h[0] = *reinterpret_cast<const v8h*>(&W2s[row * W2SS + 32 * c + 8 * hi16]);
        Bop.h[1] = *reinterpret_cast<const v8h*>(&W2s[row * W2SS + 32 * c + 16 + 8 * hi16]);
        acc2[nt] = __builtin_amdgcn_wmma_f32_16x16x32_f16(
            false, A1[c].v, false, Bop.v, (short)0, acc2[nt], false, false);
      }
    }
    // bias + relu, repack C-layout -> row-major f16 for next A operand
    compiler_fence();
#pragma unroll
    for (int nt = 0; nt < 2; ++nt) {
      float bb = b2[32 * cc + nt * 16 + m16];
#pragma unroll
      for (int r = 0; r < 8; ++r) {
        float vo = fmaxf(acc2[nt][r] + bb, 0.f);
        stg[(wv * 16 + r + 8 * hi16) * STS + nt * 16 + m16] = (half_t)vo;
      }
    }
    compiler_fence();
    V16 A2;
    A2.h[0] = *reinterpret_cast<const v8h*>(&stg[(wv * 16 + m16) * STS + 8 * hi16]);
    A2.h[1] = *reinterpret_cast<const v8h*>(&stg[(wv * 16 + m16) * STS + 16 + 8 * hi16]);
    compiler_fence();
#pragma unroll
    for (int n = 0; n < 4; ++n) {         // layer-3 N = 64
      int row = n * 16 + m16;
      V16 B3;
      B3.h[0] = *reinterpret_cast<const v8h*>(&W3s[row * W3SS + 32 * cc + 8 * hi16]);
      B3.h[1] = *reinterpret_cast<const v8h*>(&W3s[row * W3SS + 32 * cc + 16 + 8 * hi16]);
      acc3[n] = __builtin_amdgcn_wmma_f32_16x16x32_f16(
          false, A2.v, false, B3.v, (short)0, acc3[n], false, false);
    }
  }

  // layer 4: relu(acc3 + b3) . W4, reduce across the 16 N-lanes, sigmoid, * A_a
  float part[8] = {0.f, 0.f, 0.f, 0.f, 0.f, 0.f, 0.f, 0.f};
#pragma unroll
  for (int n = 0; n < 4; ++n) {
    int col = n * 16 + m16;
    float b3v = b3[col], w4v = W4[col];
#pragma unroll
    for (int r = 0; r < 8; ++r)
      part[r] += fmaxf(acc3[n][r] + b3v, 0.f) * w4v;
  }
#pragma unroll
  for (int msk = 1; msk < 16; msk <<= 1) {
#pragma unroll
    for (int r = 0; r < 8; ++r)
      part[r] += __shfl_xor(part[r], msk, 32);
  }
  if (m16 == 0) {                          // lanes 0 (rows 0-7) and 16 (rows 8-15)
    int v = v0 + wv;
    float b4v = b4[0];
#pragma unroll
    for (int r = 0; r < 8; ++r) {
      int u = u0 + r + 8 * hi16;
      float sv = 1.f / (1.f + __expf(-(part[r] + b4v)));
      wout[((size_t)b * Cn + u) * Cn + v] = sv * A_a[(size_t)u * Cn + v];
    }
  }
}

// ---------------------------------------------------------------- k3: P = w^T H, Q = (A_m^T H) .* H
__global__ void k3_pq(const float* __restrict__ h, const float* __restrict__ wmat,
                      const float* __restrict__ A_m,
                      float* __restrict__ P, float* __restrict__ Q) {
  __shared__ float Hs[Cn * Dn];            // 32 KB, one (b,t) slab
  int bt = blockIdx.x;                     // b*T + t
  int b = bt / Tn;
  size_t base = (size_t)bt * Cn * Dn;
  for (int i = threadIdx.x; i < Cn * Dn; i += blockDim.x) Hs[i] = h[base + i];
  __syncthreads();
  const float* wb = wmat + (size_t)b * Cn * Cn;
  for (int i = threadIdx.x; i < Cn * Dn; i += blockDim.x) {
    int v = i >> 6, d = i & 63;
    float pa = 0.f, pm = 0.f;
    for (int u = 0; u < Cn; ++u) {
      float hv = Hs[u * Dn + d];
      pa = fmaf(wb[(size_t)u * Cn + v], hv, pa);
      pm = fmaf(A_m[(size_t)u * Cn + v], hv, pm);
    }
    P[base + i] = pa;
    Q[base + i] = pm * Hs[v * Dn + d];
  }
}

// ---------------------------------------------------------------- k4: final combine
__global__ void k4_out(const float* __restrict__ h, const float* __restrict__ P,
                       const float* __restrict__ Q, const float* __restrict__ W_add,
                       const float* __restrict__ b_add, const float* __restrict__ W_mod,
                       const float* __restrict__ b_mod, const float* __restrict__ beta1,
                       const float* __restrict__ beta2, const float* __restrict__ beta3,
                       float* __restrict__ out) {
  size_t gid = (size_t)blockIdx.x * blockDim.x + threadIdx.x;
  size_t total = (size_t)Bn * Tn * Cn * Dn;
  if (gid >= total) return;
  int d = gid & 63;
  size_t rowbase = gid - d;
  float sa = 0.f, sm = 0.f;
  for (int k = 0; k < Dn; ++k) {
    sa = fmaf(P[rowbase + k], W_add[d * Dn + k], sa);
    sm = fmaf(Q[rowbase + k], W_mod[d * Dn + k], sm);
  }
  out[gid] = beta1[0] * h[gid] + beta2[0] * (sa + b_add[d]) + beta3[0] * (sm + b_mod[d]);
}

// ---------------------------------------------------------------- launch
extern "C" void kernel_launch(void* const* d_in, const int* in_sizes, int n_in,
                              void* d_out, int out_size, void* d_ws, size_t ws_size,
                              hipStream_t stream) {
  const float* h     = (const float*)d_in[0];
  const float* W1    = (const float*)d_in[1];
  const float* b1    = (const float*)d_in[2];
  const float* W2    = (const float*)d_in[3];
  const float* b2    = (const float*)d_in[4];
  const float* W3    = (const float*)d_in[5];
  const float* b3    = (const float*)d_in[6];
  const float* W4    = (const float*)d_in[7];
  const float* b4    = (const float*)d_in[8];
  const float* A_a   = (const float*)d_in[9];
  const float* A_m   = (const float*)d_in[10];
  const float* W_add = (const float*)d_in[11];
  const float* b_add = (const float*)d_in[12];
  const float* W_mod = (const float*)d_in[13];
  const float* b_mod = (const float*)d_in[14];
  const float* be1   = (const float*)d_in[15];
  const float* be2   = (const float*)d_in[16];
  const float* be3   = (const float*)d_in[17];

  char* ws = (char*)d_ws;
  float*  fu   = (float*)(ws + OFF_FU);
  float*  fv   = (float*)(ws + OFF_FV);
  float*  wmat = (float*)(ws + OFF_W);
  half_t* W2h  = (half_t*)(ws + OFF_W2H);
  half_t* W3h  = (half_t*)(ws + OFF_W3H);
  float*  P    = (float*)(ws + OFF_P);
  float*  Q    = (float*)(ws + OFF_Q);
  float*  out  = (float*)d_out;

  k0_convert<<<(H2 * H1 + 255) / 256, 256, 0, stream>>>(W2, W3, W2h, W3h);
  k1_fufv<<<Bn * (Cn / 8), 256, 0, stream>>>(h, W1, fu, fv);
  k2_pairmlp<<<Bn * 8 * 16, 256, K2_SMEM, stream>>>(fu, fv, b1, W2h, b2, W3h, b3, W4, b4,
                                                    A_a, wmat);
  k3_pq<<<Bn * Tn, 256, 0, stream>>>(h, wmat, A_m, P, Q);
  size_t total = (size_t)Bn * Tn * Cn * Dn;
  k4_out<<<(unsigned)((total + 255) / 256), 256, 0, stream>>>(
      h, P, Q, W_add, b_add, W_mod, b_mod, be1, be2, be3, out);
}
